// EncoderDecoder_51221779972171
// MI455X (gfx1250) — matrix operand order repaired and checked by reading, not compiled, not run on hardware
//
#include <hip/hip_runtime.h>
#include <cmath>

// ---------------------------------------------------------------------------
// Seq2seq encoder-decoder LSTM + Luong attention for MI455X (gfx1250).
// Strategy: convert/transpose all weights to bf16 once (fits in 192MB L2),
// run every GEMM through v_wmma_f32_16x16x32_bf16 with f32 accumulation.
// A-panels staged to LDS with GLOBAL_LOAD_ASYNC_TO_LDS_B128 (ASYNCcnt).
// Recurrent steps are chained via stream-ordered launches.
// ---------------------------------------------------------------------------

typedef __attribute__((ext_vector_type(8)))  __bf16 v8bf;
typedef __attribute__((ext_vector_type(16))) __bf16 v16bf;
typedef __attribute__((ext_vector_type(8)))  float  v8f;

union ABFrag { v16bf v; v8bf h[2]; };

#define BB 32      // batch
#define SS 64      // source length
#define HH 1024    // hidden
#define VV 32000   // vocab
#define TT 64      // decode steps

__device__ __forceinline__ float sigmf(float x) { return 1.0f / (1.0f + expf(-x)); }

// ---------------------------------------------------------------------------
// Generic bf16 WMMA GEMM:  C[M x N] = A[M x K] * W[K x N]  (W stored transposed
// as WT[N x K], row-major, ld = ldwt).  M = 32*gridDim.y, N = 256*gridDim.x.
// Epilogue: optional bias, optional tanh, optional f32 and/or bf16 stores.
// Block = 256 threads = 8 waves: 2 M-halves x 4 N-groups of 64 columns.
// ---------------------------------------------------------------------------
__global__ __launch_bounds__(256) void gemm_bf16_wmma(
    const __bf16* __restrict__ A, int lda,
    const __bf16* __restrict__ WT, int ldwt,
    const float* __restrict__ bias,
    float* __restrict__ Cf, long long ldc,
    __bf16* __restrict__ Cbf, int ldcbf,
    int K, int act)
{
    __shared__ alignas(64) __bf16 Alds[32 * 128];   // 8 KB A-panel chunk

    const int tid   = threadIdx.x;
    const int lane  = tid & 31;
    const int w     = tid >> 5;
    const int wm    = w & 1;          // which 16 rows
    const int wn    = w >> 1;         // which 64-col group
    const int m0    = wm << 4;
    const int nb    = blockIdx.x * 256 + wn * 64;
    const int row0  = blockIdx.y * 32;
    const int lhalf = lane >> 4;      // 0/1
    const int lmod  = lane & 15;

    v8f acc[4];
#pragma unroll
    for (int j = 0; j < 4; ++j)
        acc[j] = (v8f){0.f,0.f,0.f,0.f,0.f,0.f,0.f,0.f};

    const int sr = tid >> 3;          // staging row 0..31
    const int sc = (tid & 7) << 4;    // staging col 0..112 (16 bf16 each)

    for (int kbase = 0; kbase < K; kbase += 128) {
        if (kbase) __syncthreads();
#if defined(__gfx1250__)
        {
            // Async DMA the 32x128 bf16 A-chunk straight into LDS, bypassing
            // VGPRs.  32 bytes per lane: base + offset:16 apply to BOTH the
            // LDS and global addresses (ISA 10.7 async addressing).
            unsigned ldsa = (unsigned)(unsigned long long)&Alds[sr * 128 + sc];
            unsigned long long ga =
                (unsigned long long)(A + (size_t)(row0 + sr) * lda + kbase + sc);
            asm volatile(
                "global_load_async_to_lds_b128 %0, %1, off\n\t"
                "global_load_async_to_lds_b128 %0, %1, off offset:16\n\t"
                "s_wait_asynccnt 0"
                :: "v"(ldsa), "v"(ga) : "memory");
        }
#else
        *(v16bf*)&Alds[sr * 128 + sc] =
            *(const v16bf*)(A + (size_t)(row0 + sr) * lda + kbase + sc);
#endif
        __syncthreads();

        if (kbase + 128 < K)  // prefetch next weight chunk (global_prefetch_b8)
            __builtin_prefetch(WT + (size_t)(nb + lmod) * ldwt + kbase + 128, 0, 1);

#pragma unroll
        for (int kk = 0; kk < 4; ++kk) {
            // A fragment (16x32 bf16): lanes 0-15 hold K 0-15 of their row,
            // lanes 16-31 hold K 16-31 (matches B fragment)
            ABFrag a;
            const int kl = (kk << 5) + (lhalf << 4);
            a.h[0] = *(const v8bf*)&Alds[(m0 + lmod) * 128 + kl];
            a.h[1] = *(const v8bf*)&Alds[(m0 + lmod) * 128 + kl + 8];

            const int kg = kbase + (kk << 5) + (lhalf << 4);
#pragma unroll
            for (int j = 0; j < 4; ++j) {
                // B fragment (32x16): lane -> column, halves -> K 0-15 / 16-31.
                // WT is N x K row-major so this is one contiguous 32B read.
                const int colB = nb + (j << 4) + lmod;
                v16bf bfrag = *(const v16bf*)(WT + (size_t)colB * ldwt + kg);
                acc[j] = __builtin_amdgcn_wmma_f32_16x16x32_bf16(
                    false, a.v, false, bfrag, (short)0, acc[j], false, false);
            }
        }
    }

    // Epilogue: C layout — lanes 0-15 rows m0..m0+7, lanes 16-31 rows m0+8..m0+15
#pragma unroll
    for (int j = 0; j < 4; ++j) {
        const int col = nb + (j << 4) + lmod;
        const float bv = bias ? bias[col] : 0.f;
#pragma unroll
        for (int jj = 0; jj < 8; ++jj) {
            const int row = row0 + m0 + (lhalf << 3) + jj;
            float vv = acc[j][jj] + bv;
            if (act) vv = tanhf(vv);
            if (Cf)  Cf[(size_t)row * (size_t)ldc + col] = vv;
            if (Cbf) Cbf[(size_t)row * ldcbf + col] = (__bf16)vv;
        }
    }
}

// ---------------------------------------------------------------------------
// One-time weight convert + transpose:  WT[n*ldt + kofs + k] = bf16(W[k*Nd + n])
// ---------------------------------------------------------------------------
__global__ void cvt_transpose(const float* __restrict__ W, __bf16* __restrict__ WT,
                              int Kd, int Nd, int ldt, int kofs, long long total)
{
    long long i = (long long)blockIdx.x * blockDim.x + threadIdx.x;
    if (i >= total) return;
    int n = (int)(i / Kd);
    int k = (int)(i % Kd);
    WT[(size_t)n * ldt + kofs + k] = (__bf16)W[(size_t)k * Nd + n];
}

// ---------------------------------------------------------------------------
// Init h=c=0, xh[:,H:2H]=0, y=BOS
// ---------------------------------------------------------------------------
__global__ void init_state(float* __restrict__ h, float* __restrict__ c,
                           __bf16* __restrict__ xh, int* __restrict__ y)
{
    int idx = blockIdx.x * blockDim.x + threadIdx.x;   // B*H
    int b = idx >> 10, j = idx & 1023;
    h[idx] = 0.f; c[idx] = 0.f;
    xh[b * 2048 + 1024 + j] = (__bf16)0.f;
    if (idx < BB) y[idx] = 1;   // BOS
}

// ---------------------------------------------------------------------------
// Embedding gather -> bf16 into xh[:, 0:H]
// ---------------------------------------------------------------------------
__global__ void embed_gather(const float* __restrict__ emb, const int* __restrict__ tok,
                             int tokStride, int tokOfs, __bf16* __restrict__ xh)
{
    int idx = blockIdx.x * blockDim.x + threadIdx.x;   // B*H
    int b = idx >> 10, j = idx & 1023;
    int t = tok[b * tokStride + tokOfs];
    xh[b * 2048 + j] = (__bf16)emb[(size_t)t * HH + j];
}

// ---------------------------------------------------------------------------
// LSTM gate nonlinearity (Keras order i,f,g,o) + state update.
// Decoder mask_zero: if y[b]==0 keep old state.  Encoder: store hs slice.
// Also writes new h (bf16) into xh[:, H:2H] for the next step's fused GEMM.
// ---------------------------------------------------------------------------
__global__ void lstm_gate(const float* __restrict__ z, float* __restrict__ h,
                          float* __restrict__ c, __bf16* __restrict__ xh,
                          float* __restrict__ hs, __bf16* __restrict__ hsb,
                          const int* __restrict__ y, int t)
{
    int idx = blockIdx.x * blockDim.x + threadIdx.x;   // B*H
    int b = idx >> 10, j = idx & 1023;
    float zi = z[(b << 12) + j];
    float zf = z[(b << 12) + 1024 + j];
    float zg = z[(b << 12) + 2048 + j];
    float zo = z[(b << 12) + 3072 + j];
    float c0 = c[idx], h0 = h[idx];
    float c2 = sigmf(zf) * c0 + sigmf(zi) * tanhf(zg);
    float h2 = sigmf(zo) * tanhf(c2);
    if (y && y[b] == 0) { c2 = c0; h2 = h0; }
    h[idx] = h2; c[idx] = c2;
    xh[b * 2048 + 1024 + j] = (__bf16)h2;
    if (hs) {
        size_t o = ((size_t)b * SS + t) * HH + j;
        hs[o] = h2;
        hsb[o] = (__bf16)h2;
    }
}

// ---------------------------------------------------------------------------
// Attention scores: score[b][s] = h[b] . hs_a[b][s] , masked where source==0.
// One wave32 per (b,s); lane-strided dot + shfl_xor reduction.
// ---------------------------------------------------------------------------
__global__ void attn_score(const float* __restrict__ h, const float* __restrict__ hsa,
                           const int* __restrict__ source, float* __restrict__ score)
{
    int gw = (blockIdx.x * blockDim.x + threadIdx.x) >> 5;  // 0..B*S-1
    int lane = threadIdx.x & 31;
    int b = gw >> 6;
    const float* hp = h + b * HH;
    const float* kp = hsa + (size_t)gw * HH;
    float sum = 0.f;
    for (int k = lane; k < HH; k += 32) sum += hp[k] * kp[k];
#pragma unroll
    for (int off = 16; off; off >>= 1) sum += __shfl_xor(sum, off, 32);
    if (lane == 0)
        score[gw] = (source[gw] == 0) ? -1e9f : sum;
}

// ---------------------------------------------------------------------------
// Softmax over S=64 per batch row.
// ---------------------------------------------------------------------------
__global__ void attn_softmax(const float* __restrict__ score, float* __restrict__ att)
{
    __shared__ float sm[64];
    int b = blockIdx.x, t = threadIdx.x;
    float v = score[b * SS + t];
    sm[t] = v; __syncthreads();
    for (int o = 32; o; o >>= 1) { if (t < o) sm[t] = fmaxf(sm[t], sm[t + o]); __syncthreads(); }
    float mx = sm[0]; __syncthreads();
    float e = expf(v - mx);
    sm[t] = e; __syncthreads();
    for (int o = 32; o; o >>= 1) { if (t < o) sm[t] += sm[t + o]; __syncthreads(); }
    att[b * SS + t] = e / sm[0];
}

// ---------------------------------------------------------------------------
// ctx[b] = sum_s att[b][s] * hs[b][s]; emit cat = bf16([ctx, h]) for W_c GEMM.
// ---------------------------------------------------------------------------
__global__ void ctx_cat(const float* __restrict__ att, const float* __restrict__ hs,
                        const float* __restrict__ h, __bf16* __restrict__ cat)
{
    int idx = blockIdx.x * blockDim.x + threadIdx.x;   // B*H
    int b = idx >> 10, j = idx & 1023;
    const float* ap = att + b * SS;
    const float* hp = hs + (size_t)b * SS * HH + j;
    float acc = 0.f;
#pragma unroll 4
    for (int s = 0; s < SS; ++s) acc += ap[s] * hp[(size_t)s * HH];
    cat[b * 2048 + j]        = (__bf16)acc;
    cat[b * 2048 + 1024 + j] = (__bf16)h[idx];
}

// ---------------------------------------------------------------------------
// In-place softmax over V=32000 on this step's logits + greedy argmax -> y.
// One block per batch row.  argmax(probs) == argmax(logits).
// ---------------------------------------------------------------------------
__global__ void vocab_softmax_argmax(float* __restrict__ out, int t, int* __restrict__ y)
{
    __shared__ float sv[256];
    __shared__ int   si[256];
    int b = blockIdx.x, tid = threadIdx.x;
    float* row = out + ((size_t)b * TT + t) * VV;

    float m = -3.4e38f; int mi = 0;
    for (int v = tid; v < VV; v += 256) { float x = row[v]; if (x > m) { m = x; mi = v; } }
    sv[tid] = m; si[tid] = mi; __syncthreads();
    for (int o = 128; o; o >>= 1) {
        if (tid < o) {
            if (sv[tid + o] > sv[tid] ||
                (sv[tid + o] == sv[tid] && si[tid + o] < si[tid])) {
                sv[tid] = sv[tid + o]; si[tid] = si[tid + o];
            }
        }
        __syncthreads();
    }
    float gm = sv[0];
    if (tid == 0) y[b] = si[0];
    __syncthreads();

    float s = 0.f;
    for (int v = tid; v < VV; v += 256) s += expf(row[v] - gm);
    sv[tid] = s; __syncthreads();
    for (int o = 128; o; o >>= 1) { if (tid < o) sv[tid] += sv[tid + o]; __syncthreads(); }
    float inv = 1.f / sv[0];
    for (int v = tid; v < VV; v += 256) row[v] = expf(row[v] - gm) * inv;
}

// ---------------------------------------------------------------------------
// Host orchestration
// ---------------------------------------------------------------------------
extern "C" void kernel_launch(void* const* d_in, const int* in_sizes, int n_in,
                              void* d_out, int out_size, void* d_ws, size_t ws_size,
                              hipStream_t stream)
{
    const int*   source  = (const int*)  d_in[0];
    const float* emb_enc = (const float*)d_in[1];
    const float* Wx_e    = (const float*)d_in[2];
    const float* Wh_e    = (const float*)d_in[3];
    const float* b_e     = (const float*)d_in[4];
    const float* emb_dec = (const float*)d_in[5];
    const float* Wx_d    = (const float*)d_in[6];
    const float* Wh_d    = (const float*)d_in[7];
    const float* b_d     = (const float*)d_in[8];
    const float* W_a     = (const float*)d_in[9];
    const float* W_c     = (const float*)d_in[10];
    const float* b_c     = (const float*)d_in[11];
    const float* W_out   = (const float*)d_in[12];
    const float* b_out   = (const float*)d_in[13];
    float* out = (float*)d_out;

    // ---- carve workspace ----
    char* ws = (char*)d_ws;
    size_t off = 0;
    auto alloc = [&](size_t bytes) -> void* {
        void* p = ws + off;
        off = (off + bytes + 255) & ~(size_t)255;
        return p;
    };
    __bf16* WT2e  = (__bf16*)alloc((size_t)4096 * 2048 * 2);  // [Wx_e;Wh_e]^T
    __bf16* WT2d  = (__bf16*)alloc((size_t)4096 * 2048 * 2);  // [Wx_d;Wh_d]^T
    __bf16* WTa   = (__bf16*)alloc((size_t)1024 * 1024 * 2);  // W_a^T
    __bf16* WTc   = (__bf16*)alloc((size_t)1024 * 2048 * 2);  // W_c^T
    __bf16* WTout = (__bf16*)alloc((size_t)VV   * 1024 * 2);  // W_out^T
    float*  hs    = (float*) alloc((size_t)BB * SS * HH * 4);
    __bf16* hsb   = (__bf16*)alloc((size_t)BB * SS * HH * 2);
    float*  hsa   = (float*) alloc((size_t)BB * SS * HH * 4);
    float*  h     = (float*) alloc((size_t)BB * HH * 4);
    float*  c     = (float*) alloc((size_t)BB * HH * 4);
    float*  z     = (float*) alloc((size_t)BB * 4 * HH * 4);
    __bf16* xh    = (__bf16*)alloc((size_t)BB * 2048 * 2);
    __bf16* cat   = (__bf16*)alloc((size_t)BB * 2048 * 2);
    __bf16* htb   = (__bf16*)alloc((size_t)BB * HH * 2);
    float*  score = (float*) alloc((size_t)BB * SS * 4);
    float*  att   = (float*) alloc((size_t)BB * SS * 4);
    int*    y     = (int*)   alloc((size_t)BB * 4);

    auto cvt = [&](const float* W, __bf16* WT, int Kd, int Nd, int ldt, int kofs) {
        long long tot = (long long)Kd * Nd;
        int blocks = (int)((tot + 255) / 256);
        cvt_transpose<<<blocks, 256, 0, stream>>>(W, WT, Kd, Nd, ldt, kofs, tot);
    };

    // ---- one-time weight convert/transpose to bf16 (L2-resident thereafter) ----
    cvt(Wx_e, WT2e, 1024, 4096, 2048, 0);
    cvt(Wh_e, WT2e, 1024, 4096, 2048, 1024);
    cvt(Wx_d, WT2d, 1024, 4096, 2048, 0);
    cvt(Wh_d, WT2d, 1024, 4096, 2048, 1024);
    cvt(W_a,  WTa,  1024, 1024, 1024, 0);
    cvt(W_c,  WTc,  2048, 1024, 2048, 0);
    cvt(W_out, WTout, 1024, VV, 1024, 0);

    init_state<<<(BB * HH) / 256, 256, 0, stream>>>(h, c, xh, y);

    // ---- encoder scan ----
    for (int t = 0; t < SS; ++t) {
        embed_gather<<<(BB * HH) / 256, 256, 0, stream>>>(emb_enc, source, SS, t, xh);
        gemm_bf16_wmma<<<dim3(4096 / 256, 1), 256, 0, stream>>>(
            xh, 2048, WT2e, 2048, b_e, z, 4096LL, nullptr, 0, 2048, 0);
        lstm_gate<<<(BB * HH) / 256, 256, 0, stream>>>(z, h, c, xh, hs, hsb, nullptr, t);
    }

    // ---- attention key projection: hs_a = hs @ W_a  (M=2048, K=N=1024) ----
    gemm_bf16_wmma<<<dim3(1024 / 256, (BB * SS) / 32), 256, 0, stream>>>(
        hsb, 1024, WTa, 1024, nullptr, hsa, 1024LL, nullptr, 0, 1024, 0);

    // ---- greedy autoregressive decode ----
    for (int t = 0; t < TT; ++t) {
        embed_gather<<<(BB * HH) / 256, 256, 0, stream>>>(emb_dec, y, 1, 0, xh);
        gemm_bf16_wmma<<<dim3(4096 / 256, 1), 256, 0, stream>>>(
            xh, 2048, WT2d, 2048, b_d, z, 4096LL, nullptr, 0, 2048, 0);
        lstm_gate<<<(BB * HH) / 256, 256, 0, stream>>>(z, h, c, xh, nullptr, nullptr, y, 0);

        attn_score<<<(BB * SS * 32) / 256, 256, 0, stream>>>(h, hsa, source, score);
        attn_softmax<<<BB, 64, 0, stream>>>(score, att);
        ctx_cat<<<(BB * HH) / 256, 256, 0, stream>>>(att, hs, h, cat);

        // ht = tanh([ctx,h] @ W_c + b_c)  -> bf16 activations for vocab GEMM
        gemm_bf16_wmma<<<dim3(1024 / 256, 1), 256, 0, stream>>>(
            cat, 2048, WTc, 2048, b_c, nullptr, 0, htb, 1024, 2048, 1);

        // logits into this step's output slice: row stride = MAX_LEN*V
        gemm_bf16_wmma<<<dim3(VV / 256, 1), 256, 0, stream>>>(
            htb, 1024, WTout, 1024, b_out,
            out + (size_t)t * VV, (long long)TT * VV, nullptr, 0, 1024, 0);

        vocab_softmax_argmax<<<BB, 256, 0, stream>>>(out, t, y);
    }
}
